// myNet_7447473291581
// MI455X (gfx1250) — matrix operand order repaired and checked by reading, not compiled, not run on hardware
//
#include <hip/hip_runtime.h>
#include <hip/hip_bf16.h>
#include <math.h>

#define NB 32
#define NA 48
#define HIN 64
#define EIN 32
#define HH 64
#define HG 16
#define NCELLS 5
#define NOUT 1024  // HG*HH
#define NNODE (NB*NA)

typedef __attribute__((ext_vector_type(16))) __bf16 v16bf;
typedef __attribute__((ext_vector_type(8)))  float  v8f;

union V16 { v16bf v; uint4 q[2]; };

// round-to-nearest-even f32 -> bf16 (bit manipulation; avoids relying on fptrunc)
__device__ inline unsigned short f2bf_bits(float f) {
    union { float f; unsigned u; } x; x.f = f;
    unsigned r = x.u + 0x7FFFu + ((x.u >> 16) & 1u);
    return (unsigned short)(r >> 16);
}
__device__ inline unsigned pack2bf(float a, float b) {
    union { float f; unsigned u; } xa, xb; xa.f = a; xb.f = b;
    unsigned ra = xa.u + 0x7FFFu + ((xa.u >> 16) & 1u);
    unsigned rb = xb.u + 0x7FFFu + ((xb.u >> 16) & 1u);
    return (ra >> 16) | (rb & 0xFFFF0000u);
}

// ---------------------------------------------------------------------------
// Kernel 0: pre-pack W_e [32,1024] f32 into the bf16 WMMA B-operand register
// layout: for n-tile t, lane l, dword v: n = t*16+(l&15), kb = (l&16)?16:0,
// word = {bf16 W[kb+2v][n], bf16 W[kb+2v+1][n]}  -> Wp[(t*32+l)*8 + v]
// ---------------------------------------------------------------------------
__global__ void pack_we_kernel(const float* __restrict__ We, unsigned* __restrict__ Wp) {
    int idx = blockIdx.x * blockDim.x + threadIdx.x;  // 64*32*8 = 16384
    if (idx >= 64 * 32 * 8) return;
    int v    = idx & 7;
    int lane = (idx >> 3) & 31;
    int t    = idx >> 8;
    int n  = t * 16 + (lane & 15);
    int kb = (lane & 16) ? 16 : 0;
    int k0 = kb + 2 * v;
    Wp[idx] = pack2bf(We[k0 * NOUT + n], We[(k0 + 1) * NOUT + n]);
}

// ---------------------------------------------------------------------------
// Kernel 1: hv[b,i,:] = relu(h[b,i,:] @ W_h + b_h) * (sum_j A[b,i,j] > 0.1)
// one block per node, 64 threads (f32 VALU; 12.6 MFLOP total)
// ---------------------------------------------------------------------------
__global__ void node_mlp_kernel(const float* __restrict__ A, const float* __restrict__ h,
                                const float* __restrict__ Wh, const float* __restrict__ bh,
                                float* __restrict__ hv) {
    int node = blockIdx.x;      // 0..NNODE-1
    int t = threadIdx.x;        // 0..63
    __shared__ __align__(16) float hrow[HIN];
    hrow[t] = h[node * HIN + t];
    __syncthreads();
    const float* arow = A + (size_t)node * NA;
    float s = 0.f;
    for (int k = 0; k < NA; k++) s += arow[k];
    float mask = (s > 0.1f) ? 1.f : 0.f;
    float acc = bh[t];
    for (int k = 0; k < HIN; k++) acc = fmaf(hrow[k], Wh[k * HH + t], acc);
    hv[node * HH + t] = fmaxf(acc, 0.f) * mask;
}

// ---------------------------------------------------------------------------
// Kernel 2: fused  Esum[b,j,n] = sum_i A[b,i,j] * relu(E[b,i,j,:] @ W_e[:,n] + b_e[n])
// One block per (b,j): 128 threads = 4 waves. Each wave owns 16 n-tiles.
// A-operand = 16x32 bf16 tile of E rows (3 tiles cover i=0..47), staged in LDS.
// B-operand = pre-packed W_e tile. D (16x16 f32) post-processed per the
// documented C/D layout (VGPR v: lanes0-15 -> M=v, lanes16-31 -> M=v+8),
// i-halves merged with shfl_xor(16).
// ---------------------------------------------------------------------------
__global__ void __launch_bounds__(128)
esum_wmma_kernel(const float* __restrict__ A, const float* __restrict__ E,
                 const unsigned* __restrict__ Wp, const float* __restrict__ be,
                 float* __restrict__ Esum) {
    int blk = blockIdx.x;
    int b = blk / NA, j = blk % NA;
    __shared__ __align__(16) unsigned short Elds[NA * EIN];  // bf16 bits, row stride 64B
    __shared__ __align__(16) float Alds[NA];
    int tid = threadIdx.x;

    // stage E[b, i, j, :] rows, converting f32 -> bf16
    const float* Ebase = E + ((size_t)(b * NA) * NA + j) * EIN;  // + i*NA*EIN + e
    for (int f = tid; f < NA * EIN; f += 128) {
        int i = f >> 5, e = f & 31;
        Elds[f] = f2bf_bits(Ebase[(size_t)i * NA * EIN + e]);
    }
    if (tid < NA) Alds[tid] = A[(size_t)(b * NA + tid) * NA + j];
    __syncthreads();

    int wave = tid >> 5;
    int lane = tid & 31;
    int hl   = (lane >> 4) & 1;  // lane half
    int mrow = lane & 15;

    // Build the 3 A-operands (contiguous k order used here, not the ISA table: lane<16 holds K0-15,
    // lane>=16 holds K16-31, for row M = lane&15; matches packed B).
    V16 aop[3];
    float amask[3][8];
    int kb = hl ? 16 : 0;
    #pragma unroll
    for (int t = 0; t < 3; t++) {
        int row = t * 16 + mrow;
        aop[t].q[0] = *(const uint4*)&Elds[row * EIN + kb];
        aop[t].q[1] = *(const uint4*)&Elds[row * EIN + kb + 8];
        #pragma unroll
        for (int v = 0; v < 8; v++) amask[t][v] = Alds[t * 16 + (hl ? v + 8 : v)];
    }

    float* out = Esum + (size_t)blk * NOUT;
    for (int nt = 0; nt < 16; nt++) {
        int gt = wave * 16 + nt;                // global n-tile
        V16 bop;
        const uint4* wq = (const uint4*)(Wp + ((size_t)gt * 32 + lane) * 8);
        bop.q[0] = wq[0];
        bop.q[1] = wq[1];
        float bias = be[gt * 16 + mrow];
        float acc = 0.f;
        #pragma unroll
        for (int t = 0; t < 3; t++) {
            v8f c = {};
            c = __builtin_amdgcn_wmma_f32_16x16x32_bf16(
                    /*neg_a=*/false, aop[t].v, /*neg_b=*/false, bop.v,
                    /*c_mod=*/(short)0, c, /*reuse_a=*/false, /*reuse_b=*/false);
            #pragma unroll
            for (int v = 0; v < 8; v++) {
                float val = fmaxf(c[v] + bias, 0.f);   // relu(z + b_e)
                acc += val * amask[t][v];              // * A[b,i,j], sum over i
            }
        }
        acc += __shfl_xor(acc, 16, 32);                // merge M-halves
        if (hl == 0) out[gt * 16 + mrow] = acc;
    }
}

// ---------------------------------------------------------------------------
// Kernel 3: per-node 5-step GRU recurrence (fully node-local). One block per
// node, 192 threads (6 waves). W_ih/W_hh staged in LDS (63 KB), Esum row read
// from L2 each step. f32 VALU — 250 MFLOP total, precision-sensitive.
// ---------------------------------------------------------------------------
__global__ void __launch_bounds__(192)
gru_kernel(const float* __restrict__ Esum, const float* __restrict__ hv,
           const float* __restrict__ Wih, const float* __restrict__ bih,
           const float* __restrict__ Whh, const float* __restrict__ bhh,
           float* __restrict__ out) {
    int node = blockIdx.x;
    int t = threadIdx.x;  // 0..191
    __shared__ __align__(16) float wih_s[192 * HG];   // 12 KB
    __shared__ __align__(16) float whh_s[192 * HH];   // 48 KB
    __shared__ float hc[HH];
    __shared__ float Ml[HG];
    __shared__ float gA[192], gB[192];

    for (int f = t; f < 192 * HG; f += 192) wih_s[f] = Wih[f];
    for (int f = t; f < 192 * HH; f += 192) whh_s[f] = Whh[f];
    if (t < HH) hc[t] = hv[(size_t)node * HH + t];
    float bi = bih[t], bh2 = bhh[t];
    const float* erow = Esum + (size_t)node * NOUT;
    __syncthreads();

    for (int it = 0; it < NCELLS; it++) {
        // M[g] = Esum[node, g, :] . hc
        if (t < HG) {
            float m = 0.f;
            for (int k = 0; k < HH; k++) m = fmaf(erow[t * HH + k], hc[k], m);
            Ml[t] = m;
        }
        __syncthreads();
        // gi[k] = b_ih[k] + sum_g M[g]*W_ih[k,g];  gh[k] = b_hh[k] + sum_h hc[h]*W_hh[k,h]
        {
            float a = bi;
            #pragma unroll
            for (int g = 0; g < HG; g++) a = fmaf(Ml[g], wih_s[t * HG + g], a);
            float bb = bh2;
            for (int k = 0; k < HH; k++) bb = fmaf(hc[k], whh_s[t * HH + k], bb);
            gA[t] = a;
            gB[t] = bb;
        }
        __syncthreads();
        float hnew = 0.f;
        if (t < HH) {
            float r = 1.f / (1.f + __expf(-(gA[t] + gB[t])));
            float z = 1.f / (1.f + __expf(-(gA[HH + t] + gB[HH + t])));
            float n = tanhf(gA[2 * HH + t] + r * gB[2 * HH + t]);
            hnew = (1.f - z) * n + z * hc[t];
        }
        __syncthreads();
        if (t < HH) hc[t] = hnew;
        __syncthreads();
    }
    if (t < HH) out[(size_t)node * HH + t] = hc[t];
}

// ---------------------------------------------------------------------------
extern "C" void kernel_launch(void* const* d_in, const int* in_sizes, int n_in,
                              void* d_out, int out_size, void* d_ws, size_t ws_size,
                              hipStream_t stream) {
    (void)in_sizes; (void)n_in; (void)out_size; (void)ws_size;
    const float* A   = (const float*)d_in[0];
    const float* h   = (const float*)d_in[1];
    const float* E   = (const float*)d_in[2];
    const float* Wh  = (const float*)d_in[3];
    const float* bh  = (const float*)d_in[4];
    const float* We  = (const float*)d_in[5];
    const float* be  = (const float*)d_in[6];
    const float* Wih = (const float*)d_in[7];
    const float* bih = (const float*)d_in[8];
    const float* Whh = (const float*)d_in[9];
    const float* bhh = (const float*)d_in[10];
    float* out = (float*)d_out;

    char* ws = (char*)d_ws;
    unsigned* Wp = (unsigned*)(ws + 0);          // 64*32*8 u32   = 65536 B
    float* hv    = (float*)(ws + 65536);         // 1536*64 f32   = 393216 B
    float* Esum  = (float*)(ws + 458752);        // 1536*1024 f32 = 6291456 B
                                                 // total ~6.44 MB

    pack_we_kernel<<<64, 256, 0, stream>>>(We, Wp);
    node_mlp_kernel<<<NNODE, 64, 0, stream>>>(A, h, Wh, bh, hv);
    esum_wmma_kernel<<<NNODE, 128, 0, stream>>>(A, E, Wp, be, Esum);
    gru_kernel<<<NNODE, 192, 0, stream>>>(Esum, hv, Wih, bih, Whh, bhh, out);
}
